// AttnEncoder_31456340476109
// MI455X (gfx1250) — matrix operand order repaired and checked by reading, not compiled, not run on hardware
//
#include <hip/hip_runtime.h>
#include <hip/hip_bf16.h>

// Shapes (fixed by the reference)
#define B_  256
#define T_  64
#define N_  128
#define H_  512
#define H4_ 2048

typedef __attribute__((ext_vector_type(16))) __bf16          v16bf;
typedef __attribute__((ext_vector_type(8)))  float           v8f;
typedef __attribute__((ext_vector_type(8)))  unsigned short  v8us;
typedef __attribute__((ext_vector_type(16))) unsigned short  v16us;

__device__ __forceinline__ v8f wmma_bf16(v16us a, v16us b, v8f c) {
  return __builtin_amdgcn_wmma_f32_16x16x32_bf16(
      false, __builtin_bit_cast(v16bf, a),
      false, __builtin_bit_cast(v16bf, b),
      (short)0, c, false, false);
}

__device__ __forceinline__ unsigned short f2bf(float f) {
  unsigned u = __builtin_bit_cast(unsigned, f);
  u += 0x7fffu + ((u >> 16) & 1u);           // round-to-nearest-even
  return (unsigned short)(u >> 16);
}
__device__ __forceinline__ float fexp2(float x) { return __builtin_amdgcn_exp2f(x); }
__device__ __forceinline__ float frcp(float x)  { return __builtin_amdgcn_rcpf(x); }
__device__ __forceinline__ float sigm(float x)  { return frcp(1.f + fexp2(-1.44269504f * x)); }
__device__ __forceinline__ float tanh_f(float x) {
#if __has_builtin(__builtin_amdgcn_tanhf)
  return __builtin_amdgcn_tanhf(x);          // native V_TANH_F32 (gfx1250 trans op)
#else
  float t = fexp2(2.88539008f * x);          // exp(2x)
  return 1.f - 2.f * frcp(t + 1.f);
#endif
}

// ---------------------------------------------------------------------------
// Pack a row-major (N x K) fp32 weight into bf16 WMMA B-fragments.
// Fragment layout (ISA 7.12.2, 16-bit B 32x16): for tile (ntile,ktile),
// lane L supplies 16 bf16: element e = W[ntile*16 + L%16][ktile*32 + (L/16)*8 + (e%8) + (e/8)*16]
// Stored as frag[((ntile*KT + ktile)*32 + lane)*16 + e].
// ---------------------------------------------------------------------------
__global__ void pack_frags(const float* __restrict__ W, unsigned short* __restrict__ dst,
                           int N, int K) {
  int KT = K / 32;
  int total = (N / 16) * KT * 512;
  for (int i = blockIdx.x * blockDim.x + threadIdx.x; i < total;
       i += gridDim.x * blockDim.x) {
    int e    = i & 15;
    int lane = (i >> 4) & 31;
    int tile = i >> 9;
    int kt = tile % KT;
    int nt = tile / KT;
    int n = nt * 16 + (lane & 15);
    int k = kt * 32 + (lane >> 4) * 8 + (e & 7) + (e >> 3) * 16;
    dst[i] = f2bf(W[(size_t)n * K + k]);
  }
}

__global__ void bias_sum(const float* __restrict__ a, const float* __restrict__ b,
                         float* __restrict__ out, int n) {
  int i = blockIdx.x * blockDim.x + threadIdx.x;
  if (i < n) out[i] = a[i] + b[i];
}

// ---------------------------------------------------------------------------
// z2[b][n][s] = sum_t dx[b][t][n] * W_a2[s][t] + b_a2[s]
// One block per batch item; wave w owns n-rows [w*16, w*16+16).
// M-dim = n channel, N-dim = s, K = t (64 -> 2 WMMA k-steps).
// ---------------------------------------------------------------------------
__global__ __launch_bounds__(256) void z2_kernel(const float* __restrict__ dx,
                                                 const unsigned short* __restrict__ fWa2,
                                                 const float* __restrict__ b_a2,
                                                 float* __restrict__ z2) {
  int b    = blockIdx.x;
  int wave = threadIdx.x >> 5;
  int lane = threadIdx.x & 31;
  int mrow  = wave * 16 + (lane & 15);   // n channel this lane loads (A rows)
  int khalf = (lane >> 4) * 8;
  const float* dxb = dx + (size_t)b * T_ * N_;

  for (int st = 0; st < 4; ++st) {
    v8f acc = {};
    for (int kt = 0; kt < 2; ++kt) {
      v16us Av;
#pragma unroll
      for (int j = 0; j < 8; ++j) {
        int t0 = kt * 32 + khalf + j;
        Av[j]     = f2bf(dxb[(size_t)t0 * N_ + mrow]);
        Av[j + 8] = f2bf(dxb[(size_t)(t0 + 16) * N_ + mrow]);
      }
      v16us Bv = *(const v16us*)(fWa2 + ((size_t)((st * 2 + kt) * 32 + lane) << 4));
      acc = wmma_bf16(Av, Bv, acc);
    }
#pragma unroll
    for (int j = 0; j < 8; ++j) {
      int n = wave * 16 + (lane >> 4) * 8 + j;
      int s = st * 16 + (lane & 15);
      z2[((size_t)b * N_ + n) * T_ + s] = acc[j] + b_a2[s];
    }
  }
}

// ---------------------------------------------------------------------------
// Persistent recurrence kernel: block owns 16 batch rows, runs all 64 steps.
// ---------------------------------------------------------------------------
__global__ __launch_bounds__(256, 1) void attn_lstm(
    const float* __restrict__ dx,          // (B,T,N)
    const float* __restrict__ W_a3,        // (T,)
    const float* __restrict__ b_a3,        // (1,)
    const float* __restrict__ b_a1,        // (T,)
    const unsigned short* __restrict__ fWa1,   // frags: N=64,K=1024
    const unsigned short* __restrict__ fWih,   // frags: N=2048,K=128
    const unsigned short* __restrict__ fWhh,   // frags: N=2048,K=512
    const float* __restrict__ bias,        // (4H,) = b_ih + b_hh
    const float* __restrict__ z2,          // (B,N,T)
    float* __restrict__ out)               // (B,T,H)
{
  __shared__ alignas(16) unsigned short hs[16][1024];  // cols 0..511 = h, 512..1023 = c (bf16)
  __shared__ alignas(16) unsigned short wx[16][128];   // attention-weighted input (bf16)
  __shared__ float z1[16][64];
  __shared__ float wa3[64];
  __shared__ float ba1[64];

  const int tid  = threadIdx.x;
  const int wave = tid >> 5;
  const int lane = tid & 31;
  const int b0   = blockIdx.x * 16;
  const int mA    = lane & 15;        // A-fragment row (M within tile)
  const int khalf = (lane >> 4) * 8;  // A-fragment K-half select

  for (int i = tid; i < 16 * 1024; i += 256) (&hs[0][0])[i] = 0;
  if (tid < 64) { wa3[tid] = W_a3[tid]; ba1[tid] = b_a1[tid]; }
  const float ba3 = b_a3[0];

  float ccur[4][8];                   // c-state in WMMA C-layout registers
  float hnew[4][8];
#pragma unroll
  for (int a = 0; a < 4; ++a)
#pragma unroll
    for (int j = 0; j < 8; ++j) ccur[a][j] = 0.f;
  __syncthreads();

  for (int t = 0; t < T_; ++t) {
    // ---- z1 = [h;c] @ W_a1^T + b_a1 : waves 0..3 own one 16-wide s-tile ----
    if (wave < 4) {
      v8f acc = {};
      for (int kt = 0; kt < 32; ++kt) {
        int k0 = kt * 32;
        v8us lo = *(const v8us*)&hs[mA][k0 + khalf];
        v8us hi = *(const v8us*)&hs[mA][k0 + 16 + khalf];
        v16us Av;
#pragma unroll
        for (int j = 0; j < 8; ++j) { Av[j] = lo[j]; Av[j + 8] = hi[j]; }
        v16us Bv = *(const v16us*)(fWa1 + ((size_t)((wave * 32 + kt) * 32 + lane) << 4));
        acc = wmma_bf16(Av, Bv, acc);
      }
#pragma unroll
      for (int j = 0; j < 8; ++j) {
        int m = (lane >> 4) * 8 + j;
        int s = wave * 16 + (lane & 15);
        z1[m][s] = acc[j] + ba1[s];
      }
    }
    __syncthreads();

    // ---- fused attention: e -> softmax (shfl over 16 lanes) -> wx ----
    // Thread owns row m = tid/16 and 8 columns n0..n0+7; e stays in VGPRs.
    {
      const int m  = tid >> 4;
      const int n0 = (tid & 15) * 8;
      float ev[8];
      const float* z2p = z2 + ((size_t)(b0 + m) * N_ + n0) * T_;
#pragma unroll
      for (int j = 0; j < 8; ++j) {
        const float4* zp = (const float4*)(z2p + (size_t)j * T_);
        float s = 0.f;
#pragma unroll 4
        for (int q = 0; q < 16; ++q) {
          float4 z4 = zp[q];
          int ss = q * 4;
          s += tanh_f(z1[m][ss + 0] + z4.x) * wa3[ss + 0];
          s += tanh_f(z1[m][ss + 1] + z4.y) * wa3[ss + 1];
          s += tanh_f(z1[m][ss + 2] + z4.z) * wa3[ss + 2];
          s += tanh_f(z1[m][ss + 3] + z4.w) * wa3[ss + 3];
        }
        ev[j] = s + ba3;
      }
      // row max over this thread's 8 + the 16 lanes covering the row
      float mx = ev[0];
#pragma unroll
      for (int j = 1; j < 8; ++j) mx = fmaxf(mx, ev[j]);
#pragma unroll
      for (int off = 1; off < 16; off <<= 1) mx = fmaxf(mx, __shfl_xor(mx, off, 16));
      float sum = 0.f;
#pragma unroll
      for (int j = 0; j < 8; ++j) { ev[j] = fexp2(1.44269504f * (ev[j] - mx)); sum += ev[j]; }
#pragma unroll
      for (int off = 1; off < 16; off <<= 1) sum += __shfl_xor(sum, off, 16);
      float inv = frcp(sum);
      // wx = softmax * x_t (bf16, one b128 store)
      const float4* xp = (const float4*)(dx + ((size_t)(b0 + m) * T_ + t) * N_ + n0);
      float4 x0 = xp[0], x1 = xp[1];
      v8us w8;
      w8[0] = f2bf(ev[0] * inv * x0.x);
      w8[1] = f2bf(ev[1] * inv * x0.y);
      w8[2] = f2bf(ev[2] * inv * x0.z);
      w8[3] = f2bf(ev[3] * inv * x0.w);
      w8[4] = f2bf(ev[4] * inv * x1.x);
      w8[5] = f2bf(ev[5] * inv * x1.y);
      w8[6] = f2bf(ev[6] * inv * x1.z);
      w8[7] = f2bf(ev[7] * inv * x1.w);
      *(v8us*)&wx[m][n0] = w8;
    }
    __syncthreads();

    // ---- gates = wx@W_ih^T + h@W_hh^T + bias ; LSTM pointwise in registers ----
#pragma unroll
    for (int ht = 0; ht < 4; ++ht) {
      const int h0 = wave * 64 + ht * 16;     // this wave's hidden columns
      v8f acc[4];
#pragma unroll
      for (int g = 0; g < 4; ++g) acc[g] = {};

      // K over wx (W_ih), K=128 -> 4 k-steps
      for (int kt = 0; kt < 4; ++kt) {
        int k0 = kt * 32;
        v8us lo = *(const v8us*)&wx[mA][k0 + khalf];
        v8us hi = *(const v8us*)&wx[mA][k0 + 16 + khalf];
        v16us Av;
#pragma unroll
        for (int j = 0; j < 8; ++j) { Av[j] = lo[j]; Av[j + 8] = hi[j]; }
#pragma unroll
        for (int g = 0; g < 4; ++g) {
          int ntile = g * 32 + wave * 4 + ht;
          v16us Bv = *(const v16us*)(fWih + ((size_t)((ntile * 4 + kt) * 32 + lane) << 4));
          acc[g] = wmma_bf16(Av, Bv, acc[g]);
        }
      }
      // K over h (W_hh), K=512 -> 16 k-steps
      for (int kt = 0; kt < 16; ++kt) {
        int k0 = kt * 32;
        if (kt + 1 < 16) {
          int ntile = wave * 4 + ht;  // prefetch next K-tile of gate-i stream
          __builtin_prefetch(fWhh + ((size_t)((ntile * 16 + kt + 1) * 32 + lane) << 4), 0, 1);
        }
        v8us lo = *(const v8us*)&hs[mA][k0 + khalf];
        v8us hi = *(const v8us*)&hs[mA][k0 + 16 + khalf];
        v16us Av;
#pragma unroll
        for (int j = 0; j < 8; ++j) { Av[j] = lo[j]; Av[j + 8] = hi[j]; }
#pragma unroll
        for (int g = 0; g < 4; ++g) {
          int ntile = g * 32 + wave * 4 + ht;
          v16us Bv = *(const v16us*)(fWhh + ((size_t)((ntile * 16 + kt) * 32 + lane) << 4));
          acc[g] = wmma_bf16(Av, Bv, acc[g]);
        }
      }
      // pointwise LSTM (torch gate order i,f,g,o)
      int ncol = h0 + (lane & 15);
      float bi = bias[0 * H_ + ncol], bf = bias[1 * H_ + ncol];
      float bg = bias[2 * H_ + ncol], bo = bias[3 * H_ + ncol];
#pragma unroll
      for (int j = 0; j < 8; ++j) {
        float gi = sigm(acc[0][j] + bi);
        float gf = sigm(acc[1][j] + bf);
        float gg = tanh_f(acc[2][j] + bg);
        float go = sigm(acc[3][j] + bo);
        float cn = gf * ccur[ht][j] + gi * gg;
        ccur[ht][j] = cn;
        hnew[ht][j] = go * tanh_f(cn);
      }
    }
    __syncthreads();   // all waves done reading old h/c from hs

    // publish new h/c (bf16) for next step; stream h out (f32)
#pragma unroll
    for (int ht = 0; ht < 4; ++ht) {
      int ncol = wave * 64 + ht * 16 + (lane & 15);
#pragma unroll
      for (int j = 0; j < 8; ++j) {
        int m = (lane >> 4) * 8 + j;
        hs[m][ncol]        = f2bf(hnew[ht][j]);
        hs[m][512 + ncol]  = f2bf(ccur[ht][j]);
        out[((size_t)(b0 + m) * T_ + t) * H_ + ncol] = hnew[ht][j];
      }
    }
    __syncthreads();
  }
}

// ---------------------------------------------------------------------------
extern "C" void kernel_launch(void* const* d_in, const int* in_sizes, int n_in,
                              void* d_out, int out_size, void* d_ws, size_t ws_size,
                              hipStream_t stream) {
  const float* dx   = (const float*)d_in[0];
  const float* W_a1 = (const float*)d_in[1];
  const float* b_a1 = (const float*)d_in[2];
  const float* W_a2 = (const float*)d_in[3];
  const float* b_a2 = (const float*)d_in[4];
  const float* W_a3 = (const float*)d_in[5];
  const float* b_a3 = (const float*)d_in[6];
  const float* W_ih = (const float*)d_in[7];
  const float* W_hh = (const float*)d_in[8];
  const float* b_ih = (const float*)d_in[9];
  const float* b_hh = (const float*)d_in[10];
  (void)in_sizes; (void)n_in; (void)out_size; (void)ws_size;

  // Workspace carve-up (~11.2 MB total)
  unsigned short* fWa1 = (unsigned short*)d_ws;          // 4*32*512   = 65536
  unsigned short* fWa2 = fWa1 + 65536;                   // 4*2*512    = 4096
  unsigned short* fWih = fWa2 + 4096;                    // 128*4*512  = 262144
  unsigned short* fWhh = fWih + 262144;                  // 128*16*512 = 1048576
  float* bias = (float*)(fWhh + 1048576);                // 2048 f32
  float* z2   = bias + 2048;                             // 256*128*64 f32

  pack_frags<<<256, 256, 0, stream>>>(W_a1, fWa1, 64, 1024);
  pack_frags<<<64, 256, 0, stream>>>(W_a2, fWa2, 64, 64);
  pack_frags<<<512, 256, 0, stream>>>(W_ih, fWih, H4_, N_);
  pack_frags<<<2048, 256, 0, stream>>>(W_hh, fWhh, H4_, H_);
  bias_sum<<<8, 256, 0, stream>>>(b_ih, b_hh, bias, H4_);
  z2_kernel<<<B_, 256, 0, stream>>>(dx, fWa2, b_a2, z2);
  attn_lstm<<<B_ / 16, 256, 0, stream>>>(dx, W_a3, b_a3, b_a1,
                                         fWa1, fWih, fWhh, bias, z2,
                                         (float*)d_out);
}
